// Model_29892972380489
// MI455X (gfx1250) — compile-verified, matrix operand order checked by reading, not run on hardware
//
#include <hip/hip_runtime.h>
#include <hip/hip_bf16.h>

typedef __attribute__((ext_vector_type(16))) __bf16 v16bf;
typedef __attribute__((ext_vector_type(8)))  float  v8f;

#define LEAK (1.0f / 3.0f)
#define EPS  1e-4f

// ---------- helpers ----------
__device__ __forceinline__ unsigned short f2bf(float f) {
    unsigned u = __float_as_uint(f);
    u += 0x7fffu + ((u >> 16) & 1u);           // round-to-nearest-even
    return (unsigned short)(u >> 16);
}
__device__ __forceinline__ unsigned pack2(float a, float b) {
    return (unsigned)f2bf(a) | ((unsigned)f2bf(b) << 16);
}
__device__ __forceinline__ int lower_bound(const int* __restrict__ a, int n, int key) {
    int lo = 0, hi = n;
    while (lo < hi) { int mid = (lo + hi) >> 1; if (a[mid] < key) lo = mid + 1; else hi = mid; }
    return lo;
}
union V16 { v16bf v; unsigned u[8]; };

// ---------- keys ----------
__global__ void k_keys(const int* __restrict__ c, int N, int S, int* __restrict__ keys) {
    int i = blockIdx.x * blockDim.x + threadIdx.x;
    if (i < N) keys[i] = (c[i*3]*S + c[i*3+1])*S + c[i*3+2];
}

// ---------- weight convert + swizzle into WMMA B-fragment layout ----------
// dst element e = (((k3*KC+kc)*CT+ct)*32 + lane)*16 + i ; element i of lane:
//   k = (i/8)*16 + (lane>>4)*8 + (i%8), cin = kc*32+k, n = ct*16 + (lane&15)
__global__ void k_swz(const float* __restrict__ W, unsigned short* __restrict__ dst,
                      int K3, int CinPad, int CinSrc, int Cout, int CoutPad) {
    int total = K3 * CinPad * CoutPad;
    int CT = CoutPad >> 4, KC = CinPad >> 5;
    for (int e = blockIdx.x * blockDim.x + threadIdx.x; e < total; e += gridDim.x * blockDim.x) {
        int i    = e & 15;
        int lane = (e >> 4) & 31;
        int rest = e >> 9;
        int ct = rest % CT; rest /= CT;
        int kc = rest % KC; int k3 = rest / KC;
        int half = lane >> 4;
        int k   = ((i >> 3) << 4) + (half << 3) + (i & 7);
        int cin = (kc << 5) + k;
        int n   = (ct << 4) + (lane & 15);
        float v = 0.f;
        if (cin < CinSrc && n < Cout) v = W[((size_t)k3 * CinSrc + cin) * Cout + n];
        dst[e] = f2bf(v);
    }
}

// ---------- init conv: 7^3 submanifold, Cin=1 -> offsets on K axis (K=343 pad 352) ----------
// 32-row M tile, dual accumulators per wave, B fragment loaded once per chunk.
__global__ void k_conv_init(const float* __restrict__ feats, const int* __restrict__ keys,
                            const int* __restrict__ coords, const unsigned short* __restrict__ Wf,
                            float* __restrict__ out, int N0) {
    __shared__ unsigned sA[32][16];                 // 32 rows x 32 bf16 (as 16 uints)
    const int tid = threadIdx.x, lane = tid & 31, wave = tid >> 5;
    const int m0 = blockIdx.x * 32;
    v8f acc0 = {}, acc1 = {};
    for (int kc = 0; kc < 11; ++kc) {               // ceil(343/32)
        __syncthreads();
        // gather with fused binary search: 128 thr x 8 elems = 32 rows x 32 taps
        int row = tid >> 2, cb = (tid & 3) * 8;
        int m = m0 + row;
        float v[8];
        #pragma unroll
        for (int t = 0; t < 8; ++t) {
            int kk = kc * 32 + cb + t;
            float val = 0.f;
            if (m < N0 && kk < 343) {
                int dx = kk / 49 - 3, dy = (kk / 7) % 7 - 3, dz = kk % 7 - 3;
                int cx = coords[m*3+0] + dx, cy = coords[m*3+1] + dy, cz = coords[m*3+2] + dz;
                if ((unsigned)cx < 512u && (unsigned)cy < 512u && (unsigned)cz < 512u) {
                    int key = (cx*512 + cy)*512 + cz;
                    int g = lower_bound(keys, N0, key);
                    if (g < N0 && keys[g] == key) val = feats[g];
                }
            }
            v[t] = val;
        }
        int p = cb >> 1;
        sA[row][p]   = pack2(v[0], v[1]);
        sA[row][p+1] = pack2(v[2], v[3]);
        sA[row][p+2] = pack2(v[4], v[5]);
        sA[row][p+3] = pack2(v[6], v[7]);
        __syncthreads();
        V16 A0, A1, B;
        int arow = lane & 15, half = lane >> 4;
        #pragma unroll
        for (int j = 0; j < 8; ++j) {
            int q = ((j >> 2) << 3) + (half << 2) + (j & 3);
            A0.u[j] = sA[arow][q];
            A1.u[j] = sA[arow + 16][q];
        }
        const unsigned* bp = (const unsigned*)Wf + (((size_t)kc * 4 + wave) * 32 + lane) * 8;
        #pragma unroll
        for (int j = 0; j < 8; ++j) B.u[j] = bp[j];
        acc0 = __builtin_amdgcn_wmma_f32_16x16x32_bf16(false, A0.v, false, B.v, (short)0, acc0, false, false);
        acc1 = __builtin_amdgcn_wmma_f32_16x16x32_bf16(false, A1.v, false, B.v, (short)0, acc1, false, false);
    }
    int n  = wave * 16 + (lane & 15);
    int mb = m0 + ((lane >> 4) << 3);
    #pragma unroll
    for (int r = 0; r < 8; ++r) { int m = mb + r;      if (m < N0) out[(size_t)m * 64 + n] = acc0[r]; }
    #pragma unroll
    for (int r = 0; r < 8; ++r) { int m = mb + 16 + r; if (m < N0) out[(size_t)m * 64 + n] = acc1[r]; }
}

// ---------- generic sparse conv (gather-GEMM via WMMA) ----------
// mode 0: submanifold r=1 (27 offsets, stride 1); mode 1: down-conv stride s (s^3 offsets)
// 32-row M tile per workgroup; each wave: one 16-wide Cout tile x two 16-row M tiles.
__global__ void k_sconv(const float* __restrict__ inf, const int* __restrict__ inkeys, int Nin,
                        const int* __restrict__ oc, int Mout,
                        const unsigned short* __restrict__ Wf, float* __restrict__ out,
                        int Cin, int CT, int Cout, int K3, int stride, int S_in, int mode) {
    __shared__ int sg[32];
    __shared__ unsigned sA[32][16];
    const int tid = threadIdx.x, lane = tid & 31, wave = tid >> 5;
    const int m0 = blockIdx.x * 32;
    const int ct = blockIdx.y * 4 + wave;
    const bool wvalid = ct < CT;
    const int KC = Cin >> 5;
    v8f acc0 = {}, acc1 = {};
    for (int k3 = 0; k3 < K3; ++k3) {
        __syncthreads();
        if (tid < 32) {
            int m = m0 + tid, g = -1;
            if (m < Mout) {
                int dx, dy, dz;
                if (mode == 0) { dx = k3/9 - 1; dy = (k3/3)%3 - 1; dz = k3%3 - 1; }
                else           { dx = k3/(stride*stride); dy = (k3/stride)%stride; dz = k3%stride; }
                int cx = oc[m*3+0]*stride + dx, cy = oc[m*3+1]*stride + dy, cz = oc[m*3+2]*stride + dz;
                if ((unsigned)cx < (unsigned)S_in && (unsigned)cy < (unsigned)S_in && (unsigned)cz < (unsigned)S_in) {
                    int key = (cx*S_in + cy)*S_in + cz;
                    int lo = lower_bound(inkeys, Nin, key);
                    if (lo < Nin && inkeys[lo] == key) g = lo;
                }
            }
            sg[tid] = g;
        }
        for (int kc = 0; kc < KC; ++kc) {
            __syncthreads();
            int row = tid >> 2, cb = (tid & 3) * 8;
            int g = sg[row];
            float4 va = make_float4(0.f, 0.f, 0.f, 0.f);
            float4 vb = make_float4(0.f, 0.f, 0.f, 0.f);
            if (g >= 0) {
                const float* src = inf + (size_t)g * Cin + kc * 32 + cb;
                va = *(const float4*)src;
                vb = *(const float4*)(src + 4);
            }
            int p = cb >> 1;
            sA[row][p]   = pack2(va.x, va.y);
            sA[row][p+1] = pack2(va.z, va.w);
            sA[row][p+2] = pack2(vb.x, vb.y);
            sA[row][p+3] = pack2(vb.z, vb.w);
            __syncthreads();
            if (wvalid) {
                V16 A0, A1, B;
                int arow = lane & 15, half = lane >> 4;
                #pragma unroll
                for (int j = 0; j < 8; ++j) {
                    int q = ((j >> 2) << 3) + (half << 2) + (j & 3);
                    A0.u[j] = sA[arow][q];
                    A1.u[j] = sA[arow + 16][q];
                }
                const unsigned* bp = (const unsigned*)Wf + ((((size_t)k3 * KC + kc) * CT + ct) * 32 + lane) * 8;
                #pragma unroll
                for (int j = 0; j < 8; ++j) B.u[j] = bp[j];
                acc0 = __builtin_amdgcn_wmma_f32_16x16x32_bf16(false, A0.v, false, B.v, (short)0, acc0, false, false);
                acc1 = __builtin_amdgcn_wmma_f32_16x16x32_bf16(false, A1.v, false, B.v, (short)0, acc1, false, false);
            }
        }
    }
    if (wvalid) {
        int n = ct * 16 + (lane & 15);
        if (n < Cout) {
            int mb = m0 + ((lane >> 4) << 3);
            #pragma unroll
            for (int r = 0; r < 8; ++r) { int m = mb + r;      if (m < Mout) out[(size_t)m * Cout + n] = acc0[r]; }
            #pragma unroll
            for (int r = 0; r < 8; ++r) { int m = mb + 16 + r; if (m < Mout) out[(size_t)m * Cout + n] = acc1[r]; }
        }
    }
}

// ---------- batchnorm stats: one block per channel ----------
__global__ void k_bn_stats(const float* __restrict__ x, int N, int C,
                           const float* __restrict__ g, const float* __restrict__ b,
                           float* __restrict__ scale, float* __restrict__ shift) {
    int c = blockIdx.x;
    float s = 0.f, q = 0.f;
    for (int i = threadIdx.x; i < N; i += blockDim.x) { float v = x[(size_t)i * C + c]; s += v; q += v * v; }
    __shared__ float rs[256], rq[256];
    rs[threadIdx.x] = s; rq[threadIdx.x] = q; __syncthreads();
    for (int st = 128; st > 0; st >>= 1) {
        if ((int)threadIdx.x < st) { rs[threadIdx.x] += rs[threadIdx.x + st]; rq[threadIdx.x] += rq[threadIdx.x + st]; }
        __syncthreads();
    }
    if (threadIdx.x == 0) {
        float m = rs[0] / (float)N;
        float var = rq[0] / (float)N - m * m;
        float sc = g[c] * rsqrtf(var + EPS);
        scale[c] = sc; shift[c] = b[c] - m * sc;
    }
}

__global__ void k_bn_lrelu(float* __restrict__ x, int total, int Cmask,
                           const float* __restrict__ scale, const float* __restrict__ shift) {
    int i = blockIdx.x * blockDim.x + threadIdx.x;
    if (i < total) {
        int c = i & Cmask;
        float y = x[i] * scale[c] + shift[c];
        x[i] = y > 0.f ? y : LEAK * y;
    }
}

__global__ void k_add_lrelu(float* __restrict__ x, const float* __restrict__ t, int total) {
    int i = blockIdx.x * blockDim.x + threadIdx.x;
    if (i < total) { float y = t[i] + x[i]; x[i] = y > 0.f ? y : LEAK * y; }
}

__global__ void k_zero(float* __restrict__ p, int n) {
    int i = blockIdx.x * blockDim.x + threadIdx.x;
    if (i < n) p[i] = 0.f;
}

__global__ void k_scatter(const float* __restrict__ x4, const int* __restrict__ c4, int N4,
                          float* __restrict__ dense) {
    int i = blockIdx.x * blockDim.x + threadIdx.x;
    if (i < N4) {
        int key = (c4[i*3]*8 + c4[i*3+1])*8 + c4[i*3+2];
        dense[key]       = x4[i*2 + 0];
        dense[512 + key] = x4[i*2 + 1];
    }
}

__global__ void k_mlp(const float* __restrict__ dense, const float* __restrict__ W2,
                      const float* __restrict__ b2, const float* __restrict__ W3,
                      const float* __restrict__ b3, float* __restrict__ out) {
    float s0 = 0.f, s1 = 0.f;
    for (int i = threadIdx.x; i < 1024; i += blockDim.x) {
        float v = dense[i]; s0 += v * W2[i*2]; s1 += v * W2[i*2 + 1];
    }
    __shared__ float r0[256], r1[256];
    r0[threadIdx.x] = s0; r1[threadIdx.x] = s1; __syncthreads();
    for (int st = 128; st > 0; st >>= 1) {
        if ((int)threadIdx.x < st) { r0[threadIdx.x] += r0[threadIdx.x + st]; r1[threadIdx.x] += r1[threadIdx.x + st]; }
        __syncthreads();
    }
    if (threadIdx.x == 0) {
        float v0 = r0[0] + b2[0], v1 = r1[0] + b2[1];
        v0 = v0 > 0.f ? v0 : expm1f(v0);
        v1 = v1 > 0.f ? v1 : expm1f(v1);
        float z = v0 * W3[0] + v1 * W3[1] + b3[0];
        out[0] = 1.f / (1.f + expf(-z));
    }
}

// ---------- host ----------
extern "C" void kernel_launch(void* const* d_in, const int* in_sizes, int n_in,
                              void* d_out, int out_size, void* d_ws, size_t ws_size,
                              hipStream_t stream) {
    const float* feats0 = (const float*)d_in[0];
    const float* W_init = (const float*)d_in[1];
    const float* Wd1 = (const float*)d_in[2];  const float* gd1 = (const float*)d_in[3];  const float* bd1 = (const float*)d_in[4];
    const float* W11a = (const float*)d_in[5]; const float* g11 = (const float*)d_in[6];  const float* b11 = (const float*)d_in[7];
    const float* W11b = (const float*)d_in[8];
    const float* W12a = (const float*)d_in[9]; const float* g12 = (const float*)d_in[10]; const float* b12 = (const float*)d_in[11];
    const float* W12b = (const float*)d_in[12];
    const float* Wd2 = (const float*)d_in[13]; const float* gd2 = (const float*)d_in[14]; const float* bd2 = (const float*)d_in[15];
    const float* W21a = (const float*)d_in[16]; const float* g21 = (const float*)d_in[17]; const float* b21 = (const float*)d_in[18];
    const float* W21b = (const float*)d_in[19];
    const float* W22a = (const float*)d_in[20]; const float* g22 = (const float*)d_in[21]; const float* b22 = (const float*)d_in[22];
    const float* W22b = (const float*)d_in[23];
    const float* Wd3 = (const float*)d_in[24]; const float* gd3 = (const float*)d_in[25]; const float* bd3 = (const float*)d_in[26];
    const float* Wd4 = (const float*)d_in[27]; const float* gd4 = (const float*)d_in[28]; const float* bd4 = (const float*)d_in[29];
    const float* W2 = (const float*)d_in[30]; const float* b2 = (const float*)d_in[31];
    const float* W3 = (const float*)d_in[32]; const float* b3 = (const float*)d_in[33];
    const int* c0 = (const int*)d_in[34]; const int* c1 = (const int*)d_in[35];
    const int* c2 = (const int*)d_in[36]; const int* c3 = (const int*)d_in[37];
    const int* c4 = (const int*)d_in[38];

    const int N0 = in_sizes[34] / 3, N1 = in_sizes[35] / 3, N2 = in_sizes[36] / 3,
              N3 = in_sizes[37] / 3, N4 = in_sizes[38] / 3;

    // ---- workspace carving (256B aligned) ----
    char* base = (char*)d_ws;
    size_t off = 0;
    auto alloc = [&](size_t bytes) -> char* {
        char* p = base + off;
        off = (off + bytes + 255) & ~(size_t)255;
        return p;
    };
    int* K0 = (int*)alloc((size_t)N0 * 4);
    int* K1 = (int*)alloc((size_t)N1 * 4);
    int* K2 = (int*)alloc((size_t)N2 * 4);
    int* K3k = (int*)alloc((size_t)N3 * 4);
    auto wsz = [](int K3, int CinPad, int CoutPad) { return (size_t)K3 * CinPad * CoutPad * 2; };
    unsigned short* WFinit = (unsigned short*)alloc(wsz(1, 352, 64));
    unsigned short* WFd1   = (unsigned short*)alloc(wsz(8, 64, 64));
    unsigned short* WF11a  = (unsigned short*)alloc(wsz(27, 64, 64));
    unsigned short* WF11b  = (unsigned short*)alloc(wsz(27, 64, 64));
    unsigned short* WF12a  = (unsigned short*)alloc(wsz(27, 64, 64));
    unsigned short* WF12b  = (unsigned short*)alloc(wsz(27, 64, 64));
    unsigned short* WFd2   = (unsigned short*)alloc(wsz(8, 64, 128));
    unsigned short* WF21a  = (unsigned short*)alloc(wsz(27, 128, 128));
    unsigned short* WF21b  = (unsigned short*)alloc(wsz(27, 128, 128));
    unsigned short* WF22a  = (unsigned short*)alloc(wsz(27, 128, 128));
    unsigned short* WF22b  = (unsigned short*)alloc(wsz(27, 128, 128));
    unsigned short* WFd3   = (unsigned short*)alloc(wsz(64, 128, 64));
    unsigned short* WFd4   = (unsigned short*)alloc(wsz(64, 64, 16));
    float* B0  = (float*)alloc((size_t)N0 * 64 * 4);
    float* B1a = (float*)alloc((size_t)N1 * 64 * 4);
    float* B1b = (float*)alloc((size_t)N1 * 64 * 4);
    float* B1c = (float*)alloc((size_t)N1 * 64 * 4);
    float* B2a = (float*)alloc((size_t)N2 * 128 * 4);
    float* B2b = (float*)alloc((size_t)N2 * 128 * 4);
    float* B2c = (float*)alloc((size_t)N2 * 128 * 4);
    float* B3  = (float*)alloc((size_t)N3 * 64 * 4);
    float* B4  = (float*)alloc((size_t)N4 * 2 * 4);
    float* scaleB = (float*)alloc(128 * 4);
    float* shiftB = (float*)alloc(128 * 4);
    float* dense  = (float*)alloc(1024 * 4);
    (void)ws_size; (void)n_in; (void)out_size;

    // ---- keys ----
    k_keys<<<(N0 + 255) / 256, 256, 0, stream>>>(c0, N0, 512, K0);
    k_keys<<<(N1 + 255) / 256, 256, 0, stream>>>(c1, N1, 256, K1);
    k_keys<<<(N2 + 255) / 256, 256, 0, stream>>>(c2, N2, 128, K2);
    k_keys<<<(N3 + 255) / 256, 256, 0, stream>>>(c3, N3, 32, K3k);

    // ---- weight swizzle ----
    auto swz = [&](const float* W, unsigned short* dst, int K3, int CinPad, int CinSrc, int Cout, int CoutPad) {
        int total = K3 * CinPad * CoutPad;
        k_swz<<<(total + 255) / 256, 256, 0, stream>>>(W, dst, K3, CinPad, CinSrc, Cout, CoutPad);
    };
    swz(W_init, WFinit, 1, 352, 343, 64, 64);
    swz(Wd1, WFd1, 8, 64, 64, 64, 64);
    swz(W11a, WF11a, 27, 64, 64, 64, 64);  swz(W11b, WF11b, 27, 64, 64, 64, 64);
    swz(W12a, WF12a, 27, 64, 64, 64, 64);  swz(W12b, WF12b, 27, 64, 64, 64, 64);
    swz(Wd2, WFd2, 8, 64, 64, 128, 128);
    swz(W21a, WF21a, 27, 128, 128, 128, 128); swz(W21b, WF21b, 27, 128, 128, 128, 128);
    swz(W22a, WF22a, 27, 128, 128, 128, 128); swz(W22b, WF22b, 27, 128, 128, 128, 128);
    swz(Wd3, WFd3, 64, 128, 128, 64, 64);
    swz(Wd4, WFd4, 64, 64, 64, 2, 16);

    auto conv = [&](const float* inf, const int* keys, int Nin, const int* oc, int M,
                    const unsigned short* Wf, float* out, int Cin, int Cout, int K3,
                    int stride, int S_in, int mode) {
        int CT = (Cout + 15) / 16;
        dim3 grid((M + 31) / 32, (CT + 3) / 4);
        k_sconv<<<grid, 128, 0, stream>>>(inf, keys, Nin, oc, M, Wf, out, Cin, CT, Cout, K3, stride, S_in, mode);
    };
    auto bn = [&](float* x, int N, int C, const float* g, const float* b) {
        k_bn_stats<<<C, 256, 0, stream>>>(x, N, C, g, b, scaleB, shiftB);
        int tot = N * C;
        k_bn_lrelu<<<(tot + 255) / 256, 256, 0, stream>>>(x, tot, C - 1, scaleB, shiftB);
    };

    // ---- network ----
    k_conv_init<<<(N0 + 31) / 32, 128, 0, stream>>>(feats0, K0, c0, WFinit, B0, N0);

    conv(B0, K0, N0, c1, N1, WFd1, B1a, 64, 64, 8, 2, 512, 1);
    bn(B1a, N1, 64, gd1, bd1);

    conv(B1a, K1, N1, c1, N1, WF11a, B1b, 64, 64, 27, 1, 256, 0);
    bn(B1b, N1, 64, g11, b11);
    conv(B1b, K1, N1, c1, N1, WF11b, B1c, 64, 64, 27, 1, 256, 0);
    k_add_lrelu<<<(N1 * 64 + 255) / 256, 256, 0, stream>>>(B1a, B1c, N1 * 64);

    conv(B1a, K1, N1, c1, N1, WF12a, B1b, 64, 64, 27, 1, 256, 0);
    bn(B1b, N1, 64, g12, b12);
    conv(B1b, K1, N1, c1, N1, WF12b, B1c, 64, 64, 27, 1, 256, 0);
    k_add_lrelu<<<(N1 * 64 + 255) / 256, 256, 0, stream>>>(B1a, B1c, N1 * 64);

    conv(B1a, K1, N1, c2, N2, WFd2, B2a, 64, 128, 8, 2, 256, 1);
    bn(B2a, N2, 128, gd2, bd2);

    conv(B2a, K2, N2, c2, N2, WF21a, B2b, 128, 128, 27, 1, 128, 0);
    bn(B2b, N2, 128, g21, b21);
    conv(B2b, K2, N2, c2, N2, WF21b, B2c, 128, 128, 27, 1, 128, 0);
    k_add_lrelu<<<(N2 * 128 + 255) / 256, 256, 0, stream>>>(B2a, B2c, N2 * 128);

    conv(B2a, K2, N2, c2, N2, WF22a, B2b, 128, 128, 27, 1, 128, 0);
    bn(B2b, N2, 128, g22, b22);
    conv(B2b, K2, N2, c2, N2, WF22b, B2c, 128, 128, 27, 1, 128, 0);
    k_add_lrelu<<<(N2 * 128 + 255) / 256, 256, 0, stream>>>(B2a, B2c, N2 * 128);

    conv(B2a, K2, N2, c3, N3, WFd3, B3, 128, 64, 64, 4, 128, 1);
    bn(B3, N3, 64, gd3, bd3);

    conv(B3, K3k, N3, c4, N4, WFd4, B4, 64, 2, 64, 4, 32, 1);
    bn(B4, N4, 2, gd4, bd4);

    // ---- dense head ----
    k_zero<<<4, 256, 0, stream>>>(dense, 1024);
    k_scatter<<<(N4 + 255) / 256, 256, 0, stream>>>(B4, c4, N4, dense);
    k_mlp<<<1, 256, 0, stream>>>(dense, W2, b2, W3, b3, (float*)d_out);
}
